// PredictiveColumn_8830452760756
// MI455X (gfx1250) — matrix operand order repaired and checked by reading, not compiled, not run on hardware
//
#include <hip/hip_runtime.h>
#include <hip/hip_bf16.h>
#include <math.h>

typedef __bf16 bf16_t;
typedef __attribute__((ext_vector_type(16))) __bf16 v16bf;
typedef __attribute__((ext_vector_type(8)))  float  v8f;

#define DIM 512
#define KWIN 128
#define TAU_C 0.5f
#define ETA_X_C 0.8f
#define LAMBDA_ACT_C 1e-6f

// ---------- helpers ----------

__device__ __forceinline__ unsigned fkey(float f) {
  unsigned u = __float_as_uint(f);
  return (u & 0x80000000u) ? ~u : (u | 0x80000000u); // order-preserving key
}

// exact k-th largest of 512 keys spread 16-per-lane across a wave32,
// MSB-first radix select (32 iterations, wave-shuffle reductions).
__device__ __forceinline__ unsigned kth_largest32(const unsigned* keys, int k) {
  unsigned prefix = 0u;
  #pragma unroll 1
  for (int bit = 31; bit >= 0; --bit) {
    unsigned cand = prefix | (1u << bit);
    unsigned maskhi = 0xFFFFFFFFu << bit;
    int cnt = 0;
    #pragma unroll
    for (int i = 0; i < 16; ++i)
      cnt += ((keys[i] & maskhi) >= cand) ? 1 : 0;
    #pragma unroll
    for (int off = 16; off > 0; off >>= 1)
      cnt += __shfl_xor(cnt, off, 32);
    if (cnt >= k) prefix = cand;
  }
  return prefix;
}

// ---------- weight conversion: f32 -> bf16 (+ transpose) ----------

__global__ __launch_bounds__(256) void convert_kernel(
    const float* __restrict__ Wsnap, const float* __restrict__ V,
    const float* __restrict__ R, const float* __restrict__ Rg,
    const float* __restrict__ L,
    bf16_t* __restrict__ WsnapT_bf, bf16_t* __restrict__ Wsnap_bf,
    bf16_t* __restrict__ Vt_bf, bf16_t* __restrict__ Rt_bf,
    bf16_t* __restrict__ Rgt_bf, bf16_t* __restrict__ Lt_bf)
{
  int idx = blockIdx.x * 256 + threadIdx.x;        // 0 .. 512*512-1
  int r = idx >> 9, c = idx & 511;
  int t = c * DIM + r;
  float w = Wsnap[idx];
  Wsnap_bf[idx] = (bf16_t)w;       // used as (W_snap^T)^T for feedback GEMM
  WsnapT_bf[t]  = (bf16_t)w;       // B^T for prediction GEMM
  Vt_bf[t]  = (bf16_t)V[idx];
  Rt_bf[t]  = (bf16_t)R[idx];
  Rgt_bf[t] = (bf16_t)Rg[idx];
  Lt_bf[t]  = (bf16_t)L[idx];
}

// ---------- activation prep: kWTA-gated GELU, derivative, context ----------

__global__ __launch_bounds__(256) void prep_kernel(
    const float* __restrict__ x,  const float* __restrict__ xt1,
    const float* __restrict__ xt2, const float* __restrict__ xt3,
    const float* __restrict__ bu,
    bf16_t* __restrict__ phi_x_bf, float* __restrict__ pd,
    bf16_t* __restrict__ xc_bf, bf16_t* __restrict__ phi_xc_bf,
    bf16_t* __restrict__ bu_bf)
{
  int wave = threadIdx.x >> 5;
  int lane = threadIdx.x & 31;
  int row  = blockIdx.x * 8 + wave;
  const int base = row * DIM;

  float xv[16], xcv[16];
  unsigned kx[16], kc[16];
  #pragma unroll
  for (int i = 0; i < 16; ++i) {
    int c = lane + i * 32;
    float v  = x[base + c];
    float vc = 0.5f * xt1[base + c] + 0.3f * xt2[base + c] + 0.2f * xt3[base + c];
    xv[i] = v; xcv[i] = vc;
    kx[i] = fkey(v); kc[i] = fkey(vc);
  }
  unsigned thx = kth_largest32(kx, KWIN);
  unsigned thc = kth_largest32(kc, KWIN);

  #pragma unroll
  for (int i = 0; i < 16; ++i) {
    int c = lane + i * 32;
    float v = xv[i], vc = xcv[i];
    bool mx = kx[i] >= thx;
    bool mc = kc[i] >= thc;
    float cdf  = 0.5f * (1.0f + erff(v  * 0.70710678118f));
    float pdfv = __expf(-0.5f * v * v) * 0.39894228040f;
    float cdfc = 0.5f * (1.0f + erff(vc * 0.70710678118f));
    phi_x_bf [base + c] = (bf16_t)(mx ? (v * cdf) : 0.0f);
    pd       [base + c] = mx ? (cdf + v * pdfv) : 0.0f;
    phi_xc_bf[base + c] = (bf16_t)(mc ? (vc * cdfc) : 0.0f);
    xc_bf    [base + c] = (bf16_t)vc;
    bu_bf    [base + c] = (bf16_t)bu[base + c];
  }
}

// ---------- WMMA fragment loads (CDNA5 16-bit layouts) ----------

union FragU { v16bf v; uint4 u[2]; };

// A-matrix 16x32 bf16: lanes 0-15 row m hold K[0:8)+K[16:24); lanes 16-31 K[8:16)+K[24:32)
__device__ __forceinline__ v16bf load_fragA(const bf16_t* __restrict__ base, int lane) {
  int m = lane & 15, kg = lane >> 4;
  const bf16_t* p = base + m * DIM + kg * 8;
  FragU f;
  f.u[0] = *(const uint4*)(p);
  f.u[1] = *(const uint4*)(p + 16);
  return f.v;
}

// B-matrix 32x16 bf16, loaded from B^T (N x K row-major):
// lanes 0-15 column n hold K[0:8)+K[16:24); lanes 16-31 K[8:16)+K[24:32)
__device__ __forceinline__ v16bf load_fragB(const bf16_t* __restrict__ base, int lane) {
  int n = lane & 15, half = lane >> 4;
  const bf16_t* p = base + n * DIM + half * 8;
  FragU f;
  f.u[0] = *(const uint4*)(p);
  f.u[1] = *(const uint4*)(p + 16);
  return f.v;
}

#define WMMA_BF16(A_, B_, C_) \
  __builtin_amdgcn_wmma_f32_16x16x32_bf16(false, (A_), false, (B_), (short)0, (C_), false, false)

// ---------- fused WMMA GEMM, epilogue selected by MODE ----------
// Computes S = A[M,K] @ B[K,N] with B supplied transposed (Bt = B^T, N x K).
// MODE 0 FWD : grad  = 1.5*(S+b_in - x) + 3*(td - x) - x - lam*sign(x)
// MODE 1 PRED: err_bf = bf16(bu - (S + b_out))
// MODE 2 FB  : grad += S * phi_deriv
// MODE 3 REC : dual GEMM (A,Bt)=(phi_xc,R^T) and (A2,Bt2)=(xc,R_gate^T);
//              grad += S1 * sigmoid(S2)
// MODE 4 LAT : grad += S
template<int MODE>
__global__ __launch_bounds__(256) void gemm_kernel(
    const bf16_t* __restrict__ A,  const bf16_t* __restrict__ Bt,
    const bf16_t* __restrict__ A2, const bf16_t* __restrict__ Bt2,
    const float* __restrict__ bias,
    const float* __restrict__ x,  const float* __restrict__ td,
    const float* __restrict__ bu, const float* __restrict__ pd,
    bf16_t* __restrict__ err_out, float* __restrict__ grad)
{
  int lane = threadIdx.x & 31;
  int wave = threadIdx.x >> 5;
  int wm = wave & 3, wn = wave >> 2;
  int m0 = blockIdx.y * 128 + wm * 32;   // 32 output rows per wave
  int n0 = blockIdx.x * 64  + wn * 32;   // 32 output cols per wave

  v8f acc00 = {}, acc01 = {}, acc10 = {}, acc11 = {};
  v8f bcc00 = {}, bcc01 = {}, bcc10 = {}, bcc11 = {};

  const bf16_t* a0p = A  + m0 * DIM;
  const bf16_t* a1p = A  + (m0 + 16) * DIM;
  const bf16_t* b0p = Bt + n0 * DIM;
  const bf16_t* b1p = Bt + (n0 + 16) * DIM;

  #pragma unroll 2
  for (int k = 0; k < DIM; k += 32) {
    if (k + 32 < DIM) {  // stream next A tile through L2 (global_prefetch_b8)
      __builtin_prefetch(a0p + k + 32 + (lane & 15) * DIM, 0, 3);
    }
    v16bf a0 = load_fragA(a0p + k, lane);
    v16bf a1 = load_fragA(a1p + k, lane);
    v16bf b0 = load_fragB(b0p + k, lane);
    v16bf b1 = load_fragB(b1p + k, lane);
    acc00 = WMMA_BF16(a0, b0, acc00);
    acc01 = WMMA_BF16(a0, b1, acc01);
    acc10 = WMMA_BF16(a1, b0, acc10);
    acc11 = WMMA_BF16(a1, b1, acc11);
    if constexpr (MODE == 3) {
      v16bf c0 = load_fragA(A2 + m0 * DIM + k, lane);
      v16bf c1 = load_fragA(A2 + (m0 + 16) * DIM + k, lane);
      v16bf d0 = load_fragB(Bt2 + n0 * DIM + k, lane);
      v16bf d1 = load_fragB(Bt2 + (n0 + 16) * DIM + k, lane);
      bcc00 = WMMA_BF16(c0, d0, bcc00);
      bcc01 = WMMA_BF16(c0, d1, bcc01);
      bcc10 = WMMA_BF16(c1, d0, bcc10);
      bcc11 = WMMA_BF16(c1, d1, bcc11);
    }
  }

  // C/D layout: lanes 0-15 -> N=lane, M=v ; lanes 16-31 -> N=lane-16, M=8+v
  int ncol = lane & 15;
  int mrow = (lane >> 4) * 8;
  #pragma unroll
  for (int f = 0; f < 4; ++f) {
    const v8f& a  = (f == 0) ? acc00 : (f == 1) ? acc01 : (f == 2) ? acc10 : acc11;
    const v8f& b2 = (f == 0) ? bcc00 : (f == 1) ? bcc01 : (f == 2) ? bcc10 : bcc11;
    int mm = m0 + ((f >> 1) ? 16 : 0) + mrow;
    int nn = n0 + ((f & 1) ? 16 : 0) + ncol;
    #pragma unroll
    for (int v = 0; v < 8; ++v) {
      int idx = (mm + v) * DIM + nn;
      float s = a[v];
      if constexpr (MODE == 0) {
        float fwd = s + bias[nn];
        float xv = x[idx], tdv = td[idx];
        float sg = (xv > 0.f) ? 1.f : ((xv < 0.f) ? -1.f : 0.f);
        grad[idx] = 1.5f * (fwd - xv) + 3.0f * (tdv - xv) - xv - LAMBDA_ACT_C * sg;
      } else if constexpr (MODE == 1) {
        float p = s + bias[nn];
        err_out[idx] = (bf16_t)(bu[idx] - p);
      } else if constexpr (MODE == 2) {
        grad[idx] += s * pd[idx];
      } else if constexpr (MODE == 3) {
        float g = 1.0f / (1.0f + __expf(-b2[v]));
        grad[idx] += s * g;
      } else {
        grad[idx] += s;
      }
    }
  }
}

// ---------- global norm reduce + final update ----------

__global__ void zero_kernel(float* p) {
  if (threadIdx.x == 0 && blockIdx.x == 0) p[0] = 0.0f;
}

__global__ __launch_bounds__(256) void sumsq_kernel(
    const float* __restrict__ g, int n, float* __restrict__ out)
{
  __shared__ float sm[256];
  float s = 0.0f;
  for (int i = blockIdx.x * 256 + threadIdx.x; i < n; i += gridDim.x * 256) {
    float v = g[i];
    s += v * v;
  }
  sm[threadIdx.x] = s;
  __syncthreads();
  for (int off = 128; off > 0; off >>= 1) {
    if (threadIdx.x < off) sm[threadIdx.x] += sm[threadIdx.x + off];
    __syncthreads();
  }
  if (threadIdx.x == 0) atomicAdd(out, sm[0]);
}

__global__ __launch_bounds__(256) void final_kernel(
    const float* __restrict__ x, const float* __restrict__ grad,
    const float* __restrict__ sumsq, const int* __restrict__ step,
    float* __restrict__ out, int n)
{
  float eta  = ETA_X_C / (1.0f + 0.1f * (float)step[0]);
  float c    = TAU_C * eta;                       // dx = c * grad (pre-clip)
  float nrm  = c * sqrtf(sumsq[0]);               // ||c*grad||
  float scl  = (nrm > 1.0f) ? (1.0f / (nrm + 1e-8f)) : 1.0f;
  float coef = c * scl;
  for (int i = blockIdx.x * 256 + threadIdx.x; i < n; i += gridDim.x * 256) {
    float v = x[i] + coef * grad[i];
    out[i] = fminf(5.0f, fmaxf(-5.0f, v));
  }
}

// ---------- launch ----------

extern "C" void kernel_launch(void* const* d_in, const int* in_sizes, int n_in,
                              void* d_out, int out_size, void* d_ws, size_t ws_size,
                              hipStream_t stream) {
  const float* bu    = (const float*)d_in[0];
  const float* td    = (const float*)d_in[1];
  const float* x     = (const float*)d_in[2];
  const float* xt1   = (const float*)d_in[3];
  const float* xt2   = (const float*)d_in[4];
  const float* xt3   = (const float*)d_in[5];
  const float* V     = (const float*)d_in[6];
  const float* b_in  = (const float*)d_in[7];
  const float* Wsnap = (const float*)d_in[8];
  const float* b_out = (const float*)d_in[9];
  const float* R     = (const float*)d_in[10];
  const float* Rg    = (const float*)d_in[11];
  const float* L     = (const float*)d_in[12];
  const int*   step  = (const int*)d_in[13];

  const int B = in_sizes[2] / DIM;   // 8192
  const int n = B * DIM;

  char* w = (char*)d_ws;
  auto alloc = [&](size_t bytes) {
    char* p = w;
    w += (bytes + 255) & ~(size_t)255;
    return p;
  };
  const size_t WB = (size_t)DIM * DIM * sizeof(bf16_t);
  bf16_t* WsnapT_bf = (bf16_t*)alloc(WB);
  bf16_t* Wsnap_bf  = (bf16_t*)alloc(WB);
  bf16_t* Vt_bf     = (bf16_t*)alloc(WB);
  bf16_t* Rt_bf     = (bf16_t*)alloc(WB);
  bf16_t* Rgt_bf    = (bf16_t*)alloc(WB);
  bf16_t* Lt_bf     = (bf16_t*)alloc(WB);
  bf16_t* phi_x_bf  = (bf16_t*)alloc((size_t)n * sizeof(bf16_t));
  bf16_t* phi_xc_bf = (bf16_t*)alloc((size_t)n * sizeof(bf16_t));
  bf16_t* xc_bf     = (bf16_t*)alloc((size_t)n * sizeof(bf16_t));
  bf16_t* bu_bf     = (bf16_t*)alloc((size_t)n * sizeof(bf16_t));
  bf16_t* err_bf    = (bf16_t*)alloc((size_t)n * sizeof(bf16_t));
  float*  pd        = (float*)alloc((size_t)n * sizeof(float));
  float*  grad      = (float*)alloc((size_t)n * sizeof(float));
  float*  sumsq     = (float*)alloc(256);

  convert_kernel<<<(DIM * DIM) / 256, 256, 0, stream>>>(
      Wsnap, V, R, Rg, L, WsnapT_bf, Wsnap_bf, Vt_bf, Rt_bf, Rgt_bf, Lt_bf);

  prep_kernel<<<B / 8, 256, 0, stream>>>(
      x, xt1, xt2, xt3, bu, phi_x_bf, pd, xc_bf, phi_xc_bf, bu_bf);

  dim3 gg(DIM / 64, B / 128);
  // grad = forward/topdown/decay/act terms   (fwd = bu @ V + b_in)
  gemm_kernel<0><<<gg, 256, 0, stream>>>(bu_bf, Vt_bf, nullptr, nullptr,
      b_in, x, td, nullptr, nullptr, nullptr, grad);
  // err = bu - (phi(x) @ W_snap + b_out)
  gemm_kernel<1><<<gg, 256, 0, stream>>>(phi_x_bf, WsnapT_bf, nullptr, nullptr,
      b_out, nullptr, nullptr, bu, nullptr, err_bf, nullptr);
  // grad += (err @ W_snap^T) * phi'(x)
  gemm_kernel<2><<<gg, 256, 0, stream>>>(err_bf, Wsnap_bf, nullptr, nullptr,
      nullptr, nullptr, nullptr, nullptr, pd, nullptr, grad);
  // grad += (phi(xc) @ R) * sigmoid(xc @ R_gate)
  gemm_kernel<3><<<gg, 256, 0, stream>>>(phi_xc_bf, Rt_bf, xc_bf, Rgt_bf,
      nullptr, nullptr, nullptr, nullptr, nullptr, nullptr, grad);
  // grad += phi(x) @ L
  gemm_kernel<4><<<gg, 256, 0, stream>>>(phi_x_bf, Lt_bf, nullptr, nullptr,
      nullptr, nullptr, nullptr, nullptr, nullptr, nullptr, grad);

  zero_kernel<<<1, 64, 0, stream>>>(sumsq);
  sumsq_kernel<<<512, 256, 0, stream>>>(grad, n, sumsq);
  final_kernel<<<512, 256, 0, stream>>>(x, grad, sumsq, step, (float*)d_out, n);
}
